// TrajectoryGRU_69346541961664
// MI455X (gfx1250) — compile-verified, matrix operand order checked by reading, not run on hardware
//
#include <hip/hip_runtime.h>

// ---------------------------------------------------------------------------
// TrajectoryGRU on MI455X (gfx1250): persistent per-wave GRU recurrence.
// Each wave owns a 32-row batch tile (2 WMMA M-tiles) so every w_hh/w1
// B-fragment read from LDS feeds TWO v_wmma_f32_16x16x32_f16 ops.
// Hidden state h kept in f32 registers (16 x v8f C-tiles per wave) + f16 LDS
// mirror for A-fragments. Weights staged once into LDS (~140KB / workgroup).
// Activations: native v_tanh_f32 where available; sigmoid via tanh identity.
// ---------------------------------------------------------------------------

typedef __attribute__((ext_vector_type(16))) _Float16 v16h;
typedef __attribute__((ext_vector_type(8)))  _Float16 v8h;
typedef __attribute__((ext_vector_type(8)))  float    v8f;

#define B_SZ   16384
#define T_SZ   50
#define I_SZ   6
#define H_SZ   128
#define FUT_SZ 30

// ws / LDS weight-blob layout (bytes), all 16B aligned
#define OFF_WHH   0         // 384x128 f16          (98304 B)
#define OFF_W1    98304     // 128x128 f16          (32768 B)
#define OFF_WIH   131072    // 384x8   f16 (pad k)  ( 6144 B)
#define OFF_W2    137216    // 16x128  f16 (pad n)  ( 4096 B)
#define OFF_BIAS  141312    // 656 f32: bsum[256] bihn[128] bhhn[128] b1[128] b2p[16]
#define BLOB_SZ   143936
// per-wave LDS regions (4 waves, 32 rows each)
#define OFF_H16   143936    // 4 x 32x128 f16 (8192 B each)
#define OFF_A16   176704    // 4 x 32x128 f16
#define OFF_X16   209472    // 4 x 32x8  f16 (512 B each)
#define LDS_TOTAL 211520

union AFrag { v16h v; v8h h[2]; };

__device__ __forceinline__ v8f wmma_f16(v16h a, v16h b, v8f c) {
  return __builtin_amdgcn_wmma_f32_16x16x32_f16(false, a, false, b, (short)0, c,
                                                false, false);
}

// A/B fragment from an [N][128] f16 matrix following the documented 16-bit
// 16x32 striping: lane<16 -> K = kt*32 + {0..7, 16..23};
//                 lane>=16 -> K = kt*32 + {8..15, 24..31}.
__device__ __forceinline__ v16h load_frag128(const _Float16* W, int nt, int kt,
                                             int lane) {
  int n = nt * 16 + (lane & 15);
  int koff = (lane >> 4) * 8;
  const _Float16* p = W + n * 128 + kt * 32 + koff;
  AFrag f;
  f.h[0] = *(const v8h*)p;
  f.h[1] = *(const v8h*)(p + 16);
  return f.v;
}

// Fragment for K<=8 valid (zero-padded to K=32): only lanes<16, first chunk.
__device__ __forceinline__ v16h load_frag8(const _Float16* W, int nt, int lane) {
  v8h z = {};
  AFrag f;
  int n = nt * 16 + (lane & 15);
  f.h[0] = (lane < 16) ? *(const v8h*)(W + n * 8) : z;
  f.h[1] = z;
  return f.v;
}

#define LOG2E 1.44269504088896340736f

__device__ __forceinline__ float fexp2(float x) {
#if __has_builtin(__builtin_amdgcn_exp2f)
  return __builtin_amdgcn_exp2f(x);   // raw v_exp_f32, no denorm guards
#else
  return __builtin_exp2f(x);
#endif
}
__device__ __forceinline__ float ftanh(float x) {
#if __has_builtin(__builtin_amdgcn_tanhf)
  return __builtin_amdgcn_tanhf(x);   // native v_tanh_f32 on CDNA5
#else
  return 1.0f - 2.0f * __builtin_amdgcn_rcpf(1.0f + fexp2(2.0f * LOG2E * x));
#endif
}
// sigmoid(x) = 0.5 + 0.5*tanh(x/2): 1 trans + 2 VALU when tanh is native.
__device__ __forceinline__ float fsigmoid(float x) {
  return 0.5f + 0.5f * ftanh(0.5f * x);
}

// One fused GRU cell for this wave's 32 rows (2 M-tiles).
// hA[0..3] = A-frags of rows 0..15, hA[4..7] = rows 16..31.
// hold[jt] / hold[8+jt] hold h (f32, D-tile layout) for column block jt.
__device__ __forceinline__ void gru_step(const v16h* hA, const v16h* xA,
                                         v8f* hold, const _Float16* sWhh,
                                         const _Float16* sWih,
                                         const float* sBias, _Float16* h16,
                                         int lane) {
  int j = lane & 15;
  int rb = (lane < 16) ? 0 : 8;  // row base within a 16-row tile
#pragma unroll
  for (int jt = 0; jt < 8; ++jt) {
    int col = jt * 16 + j;
    float b_r  = sBias[col];         // b_ih + b_hh (r gate)
    float b_z  = sBias[128 + col];   // b_ih + b_hh (z gate)
    float b_in = sBias[256 + col];   // b_ih (n gate)
    float b_hn = sBias[384 + col];   // b_hh (n gate)

    // ---- r gate: seed with x@w_ih, accumulate h@w_hh (B frag reused x2) ----
    v8f zf = {};
    v16h bx = load_frag8(sWih, jt, lane);
    v8f c0 = wmma_f16(xA[0], bx, zf);
    v8f c1 = wmma_f16(xA[1], bx, zf);
#pragma unroll
    for (int kt = 0; kt < 4; ++kt) {
      v16h bw = load_frag128(sWhh, jt, kt, lane);
      c0 = wmma_f16(hA[kt], bw, c0);
      c1 = wmma_f16(hA[4 + kt], bw, c1);
    }
    v8f r0, r1;
#pragma unroll
    for (int e = 0; e < 8; ++e) {
      r0[e] = fsigmoid(c0[e] + b_r);
      r1[e] = fsigmoid(c1[e] + b_r);
    }

    // ---- z gate ----
    bx = load_frag8(sWih, 8 + jt, lane);
    c0 = wmma_f16(xA[0], bx, zf);
    c1 = wmma_f16(xA[1], bx, zf);
#pragma unroll
    for (int kt = 0; kt < 4; ++kt) {
      v16h bw = load_frag128(sWhh, 8 + jt, kt, lane);
      c0 = wmma_f16(hA[kt], bw, c0);
      c1 = wmma_f16(hA[4 + kt], bw, c1);
    }
    v8f z0, z1;
#pragma unroll
    for (int e = 0; e < 8; ++e) {
      z0[e] = fsigmoid(c0[e] + b_z);
      z1[e] = fsigmoid(c1[e] + b_z);
    }

    // ---- n gate: i_n and h_n kept separate ----
    bx = load_frag8(sWih, 16 + jt, lane);
    v8f i0 = wmma_f16(xA[0], bx, zf);
    v8f i1 = wmma_f16(xA[1], bx, zf);
    c0 = zf; c1 = zf;
#pragma unroll
    for (int kt = 0; kt < 4; ++kt) {
      v16h bw = load_frag128(sWhh, 16 + jt, kt, lane);
      c0 = wmma_f16(hA[kt], bw, c0);
      c1 = wmma_f16(hA[4 + kt], bw, c1);
    }
    v8f hn0, hn1;
#pragma unroll
    for (int e = 0; e < 8; ++e) {
      float n0 = ftanh(i0[e] + b_in + r0[e] * (c0[e] + b_hn));
      float n1 = ftanh(i1[e] + b_in + r1[e] * (c1[e] + b_hn));
      // (1-z)*n + z*h  ==  n + z*(h-n)
      float hv0 = n0 + z0[e] * (hold[jt][e] - n0);
      float hv1 = n1 + z1[e] * (hold[8 + jt][e] - n1);
      hn0[e] = hv0;
      hn1[e] = hv1;
      int row = rb + e;
      h16[row * 128 + col] = (_Float16)hv0;
      h16[(16 + row) * 128 + col] = (_Float16)hv1;
    }
    hold[jt] = hn0;
    hold[8 + jt] = hn1;
  }
}

// Decoder head: x = relu(h@w1.T + b1) @ w2.T + b2 for 32 rows.
__device__ __forceinline__ void head_step(const _Float16* sW1,
                                          const _Float16* sW2,
                                          const float* sBias, _Float16* h16,
                                          _Float16* a16, _Float16* x16,
                                          float* __restrict__ out, int b0,
                                          int step, int lane) {
  int j = lane & 15;
  int rb = (lane < 16) ? 0 : 8;
  v16h hA[8];
#pragma unroll
  for (int mt = 0; mt < 2; ++mt)
#pragma unroll
    for (int kt = 0; kt < 4; ++kt)
      hA[mt * 4 + kt] = load_frag128(h16, mt, kt, lane);

#pragma unroll
  for (int jt = 0; jt < 8; ++jt) {
    v8f c0 = {}, c1 = {};
#pragma unroll
    for (int kt = 0; kt < 4; ++kt) {
      v16h bw = load_frag128(sW1, jt, kt, lane);
      c0 = wmma_f16(hA[kt], bw, c0);
      c1 = wmma_f16(hA[4 + kt], bw, c1);
    }
    float b = sBias[512 + jt * 16 + j];
#pragma unroll
    for (int e = 0; e < 8; ++e) {
      float v0 = c0[e] + b;
      float v1 = c1[e] + b;
      v0 = (v0 > 0.0f) ? v0 : 0.0f;
      v1 = (v1 > 0.0f) ? v1 : 0.0f;
      int row = rb + e;
      a16[row * 128 + jt * 16 + j] = (_Float16)v0;
      a16[(16 + row) * 128 + jt * 16 + j] = (_Float16)v1;
    }
  }

  v16h aA[8];
#pragma unroll
  for (int mt = 0; mt < 2; ++mt)
#pragma unroll
    for (int kt = 0; kt < 4; ++kt)
      aA[mt * 4 + kt] = load_frag128(a16, mt, kt, lane);
  v8f c0 = {}, c1 = {};
#pragma unroll
  for (int kt = 0; kt < 4; ++kt) {
    v16h bw = load_frag128(sW2, 0, kt, lane);
    c0 = wmma_f16(aA[kt], bw, c0);
    c1 = wmma_f16(aA[4 + kt], bw, c1);
  }
  float b2v = sBias[640 + j];
#pragma unroll
  for (int e = 0; e < 8; ++e) {
    float v0 = c0[e] + b2v;
    float v1 = c1[e] + b2v;
    int row = rb + e;
    if (j < 8) {
      x16[row * 8 + j] = (_Float16)((j < 6) ? v0 : 0.0f);
      x16[(16 + row) * 8 + j] = (_Float16)((j < 6) ? v1 : 0.0f);
    }
    if (j < 6) {
      out[(size_t)(b0 + row) * (FUT_SZ * I_SZ) + step * I_SZ + j] = v0;
      out[(size_t)(b0 + 16 + row) * (FUT_SZ * I_SZ) + step * I_SZ + j] = v1;
    }
  }
}

__device__ __forceinline__ v16h load_x_hist(const float* __restrict__ hist,
                                            int b0, int t, int tile, int lane) {
  v8h c = {};
  AFrag f;
  f.h[1] = c;
  if (lane < 16) {
    const float* xp =
        hist + ((size_t)(b0 + tile * 16 + lane) * T_SZ + t) * I_SZ;
#pragma unroll
    for (int i = 0; i < 6; ++i) c[i] = (_Float16)xp[i];
  }
  f.h[0] = c;
  return f.v;
}

__device__ __forceinline__ v16h load_x_lds(const _Float16* X, int tile,
                                           int lane) {
  v8h z = {};
  AFrag f;
  f.h[0] =
      (lane < 16) ? *(const v8h*)(X + (tile * 16 + (lane & 15)) * 8) : z;
  f.h[1] = z;
  return f.v;
}

// ---------------------------------------------------------------------------
// Prep kernel: convert weights to f16 blob + fused biases in d_ws.
// ---------------------------------------------------------------------------
__global__ void gru_prep(const float* __restrict__ w_ih,
                         const float* __restrict__ w_hh,
                         const float* __restrict__ b_ih,
                         const float* __restrict__ b_hh,
                         const float* __restrict__ w1,
                         const float* __restrict__ b1,
                         const float* __restrict__ w2,
                         const float* __restrict__ b2,
                         unsigned char* __restrict__ ws) {
  _Float16* whh_h = (_Float16*)(ws + OFF_WHH);
  _Float16* w1_h  = (_Float16*)(ws + OFF_W1);
  _Float16* wih_h = (_Float16*)(ws + OFF_WIH);
  _Float16* w2_h  = (_Float16*)(ws + OFF_W2);
  float*    bias  = (float*)(ws + OFF_BIAS);

  int gid = blockIdx.x * blockDim.x + threadIdx.x;
  if (gid < 49152) {
    whh_h[gid] = (_Float16)w_hh[gid];
  } else if (gid < 65536) {
    int k = gid - 49152;
    w1_h[k] = (_Float16)w1[k];
  } else if (gid < 68608) {
    int k = gid - 65536;
    int n = k >> 3, kk = k & 7;
    wih_h[k] = (kk < 6) ? (_Float16)w_ih[n * 6 + kk] : (_Float16)0.0f;
  } else if (gid < 70656) {
    int k = gid - 68608;
    int n = k >> 7, kk = k & 127;
    w2_h[k] = (n < 6) ? (_Float16)w2[n * 128 + kk] : (_Float16)0.0f;
  } else if (gid < 71312) {
    int k = gid - 70656;
    if (k < 256)       bias[k] = b_ih[k] + b_hh[k];
    else if (k < 384)  bias[k] = b_ih[256 + (k - 256)];
    else if (k < 512)  bias[k] = b_hh[256 + (k - 384)];
    else if (k < 640)  bias[k] = b1[k - 512];
    else               bias[k] = ((k - 640) < 6) ? b2[k - 640] : 0.0f;
  }
}

// ---------------------------------------------------------------------------
// Main persistent kernel: 128 threads (4 waves), each wave owns 32 rows.
// ---------------------------------------------------------------------------
__global__ void __launch_bounds__(128)
gru_main(const float* __restrict__ history,
         const unsigned char* __restrict__ ws, float* __restrict__ out) {
  extern __shared__ unsigned char smem[];

  // Stage weight blob into LDS (16B vector copies).
  {
    const uint4* src = (const uint4*)ws;
    uint4* dst = (uint4*)smem;
    for (int i = threadIdx.x; i < BLOB_SZ / 16; i += blockDim.x) dst[i] = src[i];
  }
  __syncthreads();

  const _Float16* sWhh = (const _Float16*)(smem + OFF_WHH);
  const _Float16* sW1  = (const _Float16*)(smem + OFF_W1);
  const _Float16* sWih = (const _Float16*)(smem + OFF_WIH);
  const _Float16* sW2  = (const _Float16*)(smem + OFF_W2);
  const float*    sBias = (const float*)(smem + OFF_BIAS);

  int tid = threadIdx.x;
  int wave = tid >> 5;
  int lane = tid & 31;
  _Float16* h16 = (_Float16*)(smem + OFF_H16 + wave * 8192);
  _Float16* a16 = (_Float16*)(smem + OFF_A16 + wave * 8192);
  _Float16* x16 = (_Float16*)(smem + OFF_X16 + wave * 512);
  int b0 = blockIdx.x * 128 + wave * 32;

  // h0 = 0: zero the f16 mirror and the f32 register state.
  {
    v8h z = {};
    for (int i = lane; i < (32 * 128) / 8; i += 32) *(v8h*)(h16 + i * 8) = z;
  }
  v8f hold[16];
#pragma unroll
  for (int jt = 0; jt < 16; ++jt) hold[jt] = (v8f){};

  // ---- Encoder: 50 steps over history ----
#pragma clang loop unroll(disable)
  for (int t = 0; t < T_SZ; ++t) {
    v16h hA[8];
#pragma unroll
    for (int mt = 0; mt < 2; ++mt)
#pragma unroll
      for (int kt = 0; kt < 4; ++kt)
        hA[mt * 4 + kt] = load_frag128(h16, mt, kt, lane);
    v16h xA[2];
    xA[0] = load_x_hist(history, b0, t, 0, lane);
    xA[1] = load_x_hist(history, b0, t, 1, lane);
    gru_step(hA, xA, hold, sWhh, sWih, sBias, h16, lane);
  }

  // x0 = history[:, -1] -> x16 (zero-padded to 8); one row per lane.
  {
    const float* xp =
        history + ((size_t)(b0 + lane) * T_SZ + (T_SZ - 1)) * I_SZ;
#pragma unroll
    for (int i = 0; i < 8; ++i)
      x16[lane * 8 + i] = (i < 6) ? (_Float16)xp[i] : (_Float16)0.0f;
  }

  // ---- Decoder: 30 steps, GRU + head ----
#pragma clang loop unroll(disable)
  for (int s = 0; s < FUT_SZ; ++s) {
    v16h hA[8];
#pragma unroll
    for (int mt = 0; mt < 2; ++mt)
#pragma unroll
      for (int kt = 0; kt < 4; ++kt)
        hA[mt * 4 + kt] = load_frag128(h16, mt, kt, lane);
    v16h xA[2];
    xA[0] = load_x_lds(x16, 0, lane);
    xA[1] = load_x_lds(x16, 1, lane);
    gru_step(hA, xA, hold, sWhh, sWih, sBias, h16, lane);
    head_step(sW1, sW2, sBias, h16, a16, x16, out, b0, s, lane);
  }
}

// ---------------------------------------------------------------------------
extern "C" void kernel_launch(void* const* d_in, const int* in_sizes, int n_in,
                              void* d_out, int out_size, void* d_ws,
                              size_t ws_size, hipStream_t stream) {
  const float* history = (const float*)d_in[0];
  const float* w_ih = (const float*)d_in[1];
  const float* w_hh = (const float*)d_in[2];
  const float* b_ih = (const float*)d_in[3];
  const float* b_hh = (const float*)d_in[4];
  const float* w1 = (const float*)d_in[5];
  const float* b1 = (const float*)d_in[6];
  const float* w2 = (const float*)d_in[7];
  const float* b2 = (const float*)d_in[8];
  float* out = (float*)d_out;
  unsigned char* ws = (unsigned char*)d_ws;

  // Prep: f16 weight blob + fused biases into workspace.
  {
    int total = 71312;
    int blocks = (total + 255) / 256;
    gru_prep<<<blocks, 256, 0, stream>>>(w_ih, w_hh, b_ih, b_hh, w1, b1, w2,
                                         b2, ws);
  }
  // Main: 128 blocks x 128 threads (4 waves x 32 batch rows), persistent over
  // all 80 timesteps; ~207KB dynamic LDS per workgroup (CDNA5: 320KB/WG).
  gru_main<<<B_SZ / 128, 128, LDS_TOTAL, stream>>>(history, ws, out);
}
